// Att_PD_layer1_11879879544294
// MI455X (gfx1250) — matrix operand order repaired and checked by reading, not compiled, not run on hardware
//
#include <hip/hip_runtime.h>

// ---------------------------------------------------------------------------
// Types for CDNA5 WMMA (gfx1250, wave32)
// ---------------------------------------------------------------------------
typedef __bf16 v16bf __attribute__((ext_vector_type(16)));
typedef __bf16 v8bf  __attribute__((ext_vector_type(8)));
typedef float  v8f   __attribute__((ext_vector_type(8)));

#define NUM_OBJ   36
#define NHID      1024
#define KIN       2048      // V_DIM + Q_DIM
#define LDS_STRIDE 1032     // 1024 + 8: keeps 16B alignment, bank-conflict-free rows
#define NEG_SLOPE 0.01f
#define MSUB      2         // M-subtiles per workgroup (M_TILE = 32 rows)
#define NTW       4         // 16-wide N-tiles per wave (64 columns/wave)

// float -> bf16 bits, round-to-nearest-even (bitwise; avoids __bf16 cast codegen)
__device__ __forceinline__ unsigned short f2bf(float f) {
  unsigned int u = __float_as_uint(f);
  u += 0x7fffu + ((u >> 16) & 1u);
  return (unsigned short)(u >> 16);
}

__device__ __forceinline__ float lrelu(float x) { return x > 0.f ? x : NEG_SLOPE * x; }
__device__ __forceinline__ float sigm(float x)  { return 1.f / (1.f + __expf(-x)); }

// ---------------------------------------------------------------------------
// Kernel 1: pack a row-major f32 weight [K,N] into bf16 WMMA B-fragments.
// Fragment tile = 32(K) x 16(N).  B layout used here (16-bit, 32x16, contiguous k):
//   lane L holds column N = L&15 ; element e holds K = (L>>4)*16 + e.
// Storage: dst[((ntg*(K/32) + kc)*32 + lane)*16 + e]  -> 32B contiguous/lane.
// ---------------------------------------------------------------------------
__global__ void pack_bf16_frag(const float* __restrict__ W,
                               unsigned short* __restrict__ dst,
                               int K, int N) {
  int gid = blockIdx.x * blockDim.x + threadIdx.x;
  int total = (K >> 5) * (N >> 4) * 32;
  if (gid >= total) return;
  int lane = gid & 31;
  int tile = gid >> 5;
  int KC   = K >> 5;
  int ntg  = tile / KC;
  int kc   = tile - ntg * KC;
  const float* src = W + (size_t)(kc * 32 + (lane >> 4) * 16) * N
                       + ntg * 16 + (lane & 15);
  unsigned short* o = dst + (size_t)gid * 16;
#pragma unroll
  for (int e = 0; e < 16; ++e) o[e] = f2bf(src[(size_t)e * N]);
}

// ---------------------------------------------------------------------------
// Kernel 2: fused dual-MLP producing logits[M] (M = T*36).
// One workgroup = 16 waves (512 thr) per 32 GEMM rows (2 A-subtiles).
// Wave w owns hidden columns [w*64, w*64+64).  Each B fragment is reused by
// both A-subtiles from registers -> halves L2 weight traffic vs 16-row tiles.
// bf16 WMMA, f32 accumulate.
// ---------------------------------------------------------------------------
__global__ void __launch_bounds__(512)
fused_mlp(const float* __restrict__ v, const float* __restrict__ q,
          const unsigned short* __restrict__ pW1w,
          const unsigned short* __restrict__ pWg1w,
          const unsigned short* __restrict__ pW2w,
          const unsigned short* __restrict__ pWg2w,
          const float* __restrict__ b1,  const float* __restrict__ bg1,
          const float* __restrict__ b2,  const float* __restrict__ bg2,
          const float* __restrict__ wlin, const float* __restrict__ blin,
          float* __restrict__ outLog, int M) {
  extern __shared__ char smem[];
  unsigned short* sH = (unsigned short*)smem;                 // [32][LDS_STRIDE] bf16
  unsigned short* sG = sH + 32 * LDS_STRIDE;                  // [32][LDS_STRIDE] bf16
  __shared__ float sLog[32];

  const v16bf* pW1  = (const v16bf*)pW1w;
  const v16bf* pWg1 = (const v16bf*)pWg1w;
  const v16bf* pW2  = (const v16bf*)pW2w;
  const v16bf* pWg2 = (const v16bf*)pWg2w;

  const int tid  = threadIdx.x;
  const int lane = tid & 31;
  const int wv   = tid >> 5;         // wave 0..15
  const int col  = lane & 15;        // N (C/D layout) / M row (A layout)
  const int half = lane >> 4;
  const int n0w  = wv * (NTW * 16);  // 64 columns per wave

  if (tid < 32) sLog[tid] = 0.f;

  // GEMM rows: r = blockIdx*32 + s*16 + col ; v is [T,36,1024] flat so
  // v-row base = r*1024. Tail workgroup clamps loads, guards stores.
  const float* vrow[MSUB];
  const float* qrow[MSUB];
#pragma unroll
  for (int s = 0; s < MSUB; ++s) {
    long r = (long)blockIdx.x * 32 + s * 16 + col;
    if (r > M - 1) r = M - 1;                    // clamp (load-safe)
    const int t = (int)(r / NUM_OBJ);
    vrow[s] = v + r * 1024;
    qrow[s] = q + (long)t * 1024;
  }

  // ---------------- Stage A: VQ[32,2048] @ {W1,Wg1} -> H1,G1 ----------------
  v8f accH[MSUB][NTW] = {}, accG[MSUB][NTW] = {};
  for (int kc = 0; kc < KIN / 32; ++kc) {
    const int kb = kc * 32;
    union { v16bf vv; unsigned short us[16]; } a[MSUB];
#pragma unroll
    for (int s = 0; s < MSUB; ++s) {
      const float* src = (kb < 1024) ? (vrow[s] + kb) : (qrow[s] + (kb - 1024));
      // 16-bit A layout (matches packed B): elem e -> K = half*16 + e
      const float4 f0 = *(const float4*)(src + half * 16);
      const float4 f1 = *(const float4*)(src + half * 16 + 4);
      const float4 f2 = *(const float4*)(src + half * 16 + 8);
      const float4 f3 = *(const float4*)(src + half * 16 + 12);
      a[s].us[0]=f2bf(f0.x);  a[s].us[1]=f2bf(f0.y);  a[s].us[2]=f2bf(f0.z);  a[s].us[3]=f2bf(f0.w);
      a[s].us[4]=f2bf(f1.x);  a[s].us[5]=f2bf(f1.y);  a[s].us[6]=f2bf(f1.z);  a[s].us[7]=f2bf(f1.w);
      a[s].us[8]=f2bf(f2.x);  a[s].us[9]=f2bf(f2.y);  a[s].us[10]=f2bf(f2.z); a[s].us[11]=f2bf(f2.w);
      a[s].us[12]=f2bf(f3.x); a[s].us[13]=f2bf(f3.y); a[s].us[14]=f2bf(f3.z); a[s].us[15]=f2bf(f3.w);
    }
#pragma unroll
    for (int nt = 0; nt < NTW; ++nt) {
      const int ntg = wv * NTW + nt;
      const v16bf bH = pW1 [(ntg * (KIN / 32) + kc) * 32 + lane];
      const v16bf bG = pWg1[(ntg * (KIN / 32) + kc) * 32 + lane];
#pragma unroll
      for (int s = 0; s < MSUB; ++s) {   // B fragment reused across subtiles
        accH[s][nt] = __builtin_amdgcn_wmma_f32_16x16x32_bf16(
            false, a[s].vv, false, bH, (short)0, accH[s][nt], false, false);
        accG[s][nt] = __builtin_amdgcn_wmma_f32_16x16x32_bf16(
            false, a[s].vv, false, bG, (short)0, accG[s][nt], false, false);
      }
    }
  }

  // Activations -> LDS (bf16), layout [row 0..31][hidden col]
#pragma unroll
  for (int nt = 0; nt < NTW; ++nt) {
    const int nc = n0w + nt * 16 + col;
    const float b1v = b1[nc], bg1v = bg1[nc];
#pragma unroll
    for (int s = 0; s < MSUB; ++s) {
#pragma unroll
      for (int j = 0; j < 8; ++j) {
        const int row = s * 16 + j + half * 8;
        sH[row * LDS_STRIDE + nc] = f2bf(lrelu(accH[s][nt][j] + b1v));
        sG[row * LDS_STRIDE + nc] = f2bf(sigm (accG[s][nt][j] + bg1v));
      }
    }
  }
  __syncthreads();

  // ---------------- Stage B: H1 @ W2 , G1 @ Wg2 ----------------
  v8f acc2H[MSUB][NTW] = {}, acc2G[MSUB][NTW] = {};
  for (int kc = 0; kc < NHID / 32; ++kc) {
    const int off0 = kc * 32 + half * 16;
    const int off1 = off0 + 8;
    v16bf aH[MSUB], aG[MSUB];
#pragma unroll
    for (int s = 0; s < MSUB; ++s) {
      const int rowb = (s * 16 + col) * LDS_STRIDE;   // A-operand row = lane&15
      const v8bf h0 = *(const v8bf*)(sH + rowb + off0);
      const v8bf h1 = *(const v8bf*)(sH + rowb + off1);
      const v8bf g0 = *(const v8bf*)(sG + rowb + off0);
      const v8bf g1 = *(const v8bf*)(sG + rowb + off1);
      aH[s] = __builtin_shufflevector(h0, h1, 0,1,2,3,4,5,6,7,8,9,10,11,12,13,14,15);
      aG[s] = __builtin_shufflevector(g0, g1, 0,1,2,3,4,5,6,7,8,9,10,11,12,13,14,15);
    }
#pragma unroll
    for (int nt = 0; nt < NTW; ++nt) {
      const int ntg = wv * NTW + nt;
      const v16bf bH = pW2 [(ntg * (NHID / 32) + kc) * 32 + lane];
      const v16bf bG = pWg2[(ntg * (NHID / 32) + kc) * 32 + lane];
#pragma unroll
      for (int s = 0; s < MSUB; ++s) {
        acc2H[s][nt] = __builtin_amdgcn_wmma_f32_16x16x32_bf16(
            false, aH[s], false, bH, (short)0, acc2H[s][nt], false, false);
        acc2G[s][nt] = __builtin_amdgcn_wmma_f32_16x16x32_bf16(
            false, aG[s], false, bG, (short)0, acc2G[s][nt], false, false);
      }
    }
  }

  // logits: (lrelu(h2)*sigm(g2)) . w_lin  -> LDS float atomics, rows 0..31
  float part[MSUB][8] = {};
#pragma unroll
  for (int nt = 0; nt < NTW; ++nt) {
    const int nc = n0w + nt * 16 + col;
    const float wl = wlin[nc], b2v = b2[nc], bg2v = bg2[nc];
#pragma unroll
    for (int s = 0; s < MSUB; ++s) {
#pragma unroll
      for (int j = 0; j < 8; ++j) {
        const float h = lrelu(acc2H[s][nt][j] + b2v);
        const float g = sigm (acc2G[s][nt][j] + bg2v);
        part[s][j] += h * g * wl;
      }
    }
  }
#pragma unroll
  for (int s = 0; s < MSUB; ++s)
#pragma unroll
    for (int j = 0; j < 8; ++j)
      atomicAdd(&sLog[s * 16 + j + half * 8], part[s][j]);
  __syncthreads();
  if (tid < 32) {
    const long r = (long)blockIdx.x * 32 + tid;
    if (r < M) outLog[r] = sLog[tid] + blin[0];
  }
}

// ---------------------------------------------------------------------------
// Kernel 3: ragged pad-scatter + allennlp masked softmax.
// out[b,g,pos,k] over B=16,G=4,ML=16,K=36. Padded rows use vec=0 (matches
// the reference's zero-scatter -> uniform-over-valid-boxes behavior).
// ---------------------------------------------------------------------------
__global__ void pad_masked_softmax(const float* __restrict__ lg,
                                   const int* __restrict__ ta,      // [16,4,16]
                                   const float* __restrict__ bm,    // [16,36]
                                   float* __restrict__ out) {
  int idx = blockIdx.x * blockDim.x + threadIdx.x;
  if (idx >= 16 * 4 * 16) return;
  const int b = idx >> 6, pos = idx & 15;
  const int seg = idx >> 4;          // b*4 + g
  int start = 0;
  for (int s = 0; s < seg; ++s)
    for (int ml = 0; ml < 16; ++ml) start += ta[s * 16 + ml];
  int len = 0;
  for (int ml = 0; ml < 16; ++ml) len += ta[seg * 16 + ml];

  float x[36];
  float mx = -3.4e38f;
  const bool valid = pos < len;
  const long trow = (long)(start + pos) * NUM_OBJ;
#pragma unroll 4
  for (int k = 0; k < NUM_OBJ; ++k) {
    const float vk = valid ? lg[trow + k] : 0.f;
    x[k] = vk * bm[b * NUM_OBJ + k];
    mx = fmaxf(mx, x[k]);
  }
  float s = 0.f;
#pragma unroll 4
  for (int k = 0; k < NUM_OBJ; ++k) { x[k] = __expf(x[k] - mx); s += x[k]; }
  const float inv = 1.f / s;
  float s2 = 0.f;
#pragma unroll 4
  for (int k = 0; k < NUM_OBJ; ++k) {
    x[k] = x[k] * inv * bm[b * NUM_OBJ + k];
    s2 += x[k];
  }
  const float inv2 = 1.f / (s2 + 1e-13f);
#pragma unroll 4
  for (int k = 0; k < NUM_OBJ; ++k) out[(long)idx * NUM_OBJ + k] = x[k] * inv2;
}

// ---------------------------------------------------------------------------
extern "C" void kernel_launch(void* const* d_in, const int* in_sizes, int n_in,
                              void* d_out, int out_size, void* d_ws, size_t ws_size,
                              hipStream_t stream) {
  const float* v    = (const float*)d_in[0];
  const float* q    = (const float*)d_in[1];
  const float* bm   = (const float*)d_in[2];
  const int*   ta   = (const int*)  d_in[3];
  const float* W1   = (const float*)d_in[4];
  const float* b1   = (const float*)d_in[5];
  const float* W2   = (const float*)d_in[6];
  const float* b2   = (const float*)d_in[7];
  const float* Wg1  = (const float*)d_in[8];
  const float* bg1  = (const float*)d_in[9];
  const float* Wg2  = (const float*)d_in[10];
  const float* bg2  = (const float*)d_in[11];
  const float* wlin = (const float*)d_in[12];
  const float* blin = (const float*)d_in[13];

  const int T = in_sizes[1] / 1024;
  const int M = T * NUM_OBJ;                 // 27504

  // d_ws layout (bf16-packed weights resident in L2, then logits)
  unsigned short* pW1  = (unsigned short*)d_ws;
  unsigned short* pWg1 = pW1  + (size_t)KIN  * NHID;
  unsigned short* pW2  = pWg1 + (size_t)KIN  * NHID;
  unsigned short* pWg2 = pW2  + (size_t)NHID * NHID;
  float* logits = (float*)(pWg2 + (size_t)NHID * NHID);

  // Pack weights (f32 -> bf16 B-fragment order)
  pack_bf16_frag<<<512, 256, 0, stream>>>(W1,  pW1,  KIN,  NHID);
  pack_bf16_frag<<<512, 256, 0, stream>>>(Wg1, pWg1, KIN,  NHID);
  pack_bf16_frag<<<256, 256, 0, stream>>>(W2,  pW2,  NHID, NHID);
  pack_bf16_frag<<<256, 256, 0, stream>>>(Wg2, pWg2, NHID, NHID);

  // Fused dual-MLP -> logits (32 rows per workgroup, tail-safe)
  const int grid = (M + 31) / 32;
  const size_t shbytes = (size_t)2 * 32 * LDS_STRIDE * sizeof(unsigned short);
  fused_mlp<<<grid, 512, shbytes, stream>>>(
      v, q, pW1, pWg1, pW2, pWg2, b1, bg1, b2, bg2, wlin, blin, logits, M);

  // Ragged pad + masked softmax -> d_out [16,4,16,36]
  pad_masked_softmax<<<8, 128, 0, stream>>>(logits, ta, bm, (float*)d_out);
}
